// FeedForwardQuantum_65481071399183
// MI455X (gfx1250) — matrix operand order repaired and checked by reading, not compiled, not run on hardware
//
#include <hip/hip_runtime.h>
#include <hip/hip_bf16.h>

typedef __attribute__((ext_vector_type(16))) _Float16 v16h;
typedef __attribute__((ext_vector_type(8)))  _Float16 v8h;
typedef __attribute__((ext_vector_type(4)))  _Float16 v4h;
typedef __attribute__((ext_vector_type(8)))  float    v8f;
typedef __attribute__((ext_vector_type(4)))  float    v4f;

#define M_ROWS 16384   // B*S = 8*2048
#define E_DIM  512
#define F_DIM  2048
#define NQ     8

// ---------------------------------------------------------------------------
// Kernel 0: convert W2 [E=512, F=2048] f32 -> f16 into workspace (2 MB).
// One-time 4 MB read + 2 MB write; W2h then stays L2-resident for the GEMM.
// ---------------------------------------------------------------------------
__global__ __launch_bounds__(256) void w2_cvt_kernel(const float* __restrict__ W2,
                                                     _Float16* __restrict__ W2h) {
  int i = (blockIdx.x * blockDim.x + threadIdx.x) * 4;
  v4f v = *(const v4f*)(W2 + i);
  v4h h;
  h[0] = (_Float16)v[0];
  h[1] = (_Float16)v[1];
  h[2] = (_Float16)v[2];
  h[3] = (_Float16)v[3];
  *(v4h*)(W2h + i) = h;
}

// ---------------------------------------------------------------------------
// Fused kernel: per 32-row stripe, compute q = cos(x[:, :8]) * cos(theta)
// once (kept in registers), then loop K in steps of 32: build the 32x32 fp16
// H tile (relu(q @ W1^T + b1)) in a DOUBLE-BUFFERED LDS tile and run
// v_wmma_f32_16x16x32_f16 against fp16 W2. H never touches DRAM
// (saves ~134 MB of HBM traffic); one barrier per K iteration.
// Block = 256 threads = 8 waves; wave w owns cols [w*64, w*64+64):
// 2 M-tiles x 4 N-tiles = 8 WMMA accumulators (64 f32 VGPRs).
// ---------------------------------------------------------------------------
__global__ __launch_bounds__(256, 1) void ffq_fused_kernel(
    const float* __restrict__ x, const float* __restrict__ theta,
    const float* __restrict__ W1, const float* __restrict__ b1,
    const float* __restrict__ b2, const _Float16* __restrict__ W2h,
    float* __restrict__ out) {
  __shared__ __align__(16) float    qs[32][NQ];
  __shared__ __align__(16) _Float16 Hs[2][32][40];  // double-buffered 32x32 tile,
                                                    // padded rows (80B, 16B-aligned)
  const int t  = threadIdx.x;
  const int m0 = blockIdx.x * 32;

  // --- Stage 1: q[r][qi] = cos(x[m0+r, qi]) * cos(theta[qi]); one value/thread
  {
    int r  = t >> 3;   // 0..31
    int qi = t & 7;    // 0..7
    float xv = x[(size_t)(m0 + r) * E_DIM + qi];
    qs[r][qi] = cosf(xv) * cosf(theta[qi]);
  }
  __syncthreads();

  // --- H-tile work split: thread computes column kk = t&31, rows rb..rb+3.
  //     Pull this thread's 4 q-rows into registers once (loop-invariant).
  const int kkid = t & 31;
  const int rb   = (t >> 5) * 4;
  float qreg[4][NQ];
#pragma unroll
  for (int rr = 0; rr < 4; ++rr)
#pragma unroll
    for (int qi = 0; qi < NQ; ++qi) qreg[rr][qi] = qs[rb + rr][qi];
  // qs is never written again -> no extra barrier needed.

  const int lane    = t & 31;
  const int wave    = t >> 5;        // 0..7
  const int lm      = lane & 15;     // row/col within tile
  const int lk      = lane >> 4;     // half-wave selector
  const int colbase = wave * 64;     // wave's 64-column slab

  v8f acc[2][4] = {};                // 2 M-tiles x 4 N-tiles, f32 accum

  int buf = 0;
  for (int k0 = 0; k0 < F_DIM; k0 += 32, buf ^= 1) {
    // --- Build Hs[buf][32][32] = relu(q @ W1[k0:k0+32]^T + b1) in fp16
    {
      const float* w1p = W1 + (size_t)(k0 + kkid) * NQ;
      float w1r[NQ];
#pragma unroll
      for (int qi = 0; qi < NQ; ++qi) w1r[qi] = w1p[qi];
      float bb = b1[k0 + kkid];
#pragma unroll
      for (int rr = 0; rr < 4; ++rr) {
        float s = bb;
#pragma unroll
        for (int qi = 0; qi < NQ; ++qi) s = fmaf(qreg[rr][qi], w1r[qi], s);
        Hs[buf][rb + rr][kkid] = (_Float16)fmaxf(s, 0.0f);
      }
    }
    __syncthreads();  // single barrier per iteration (double-buffered tile)

    // --- A fragments from LDS (ISA A-layout: lane holds row lm, K chunks at
    //     lk*8 and 16+lk*8), B fragments from fp16 W2 in global (K-contiguous)
    v16h af[2], bf[4];
#pragma unroll
    for (int mt = 0; mt < 2; ++mt) {
      const _Float16* hp = &Hs[buf][mt * 16 + lm][lk * 8];
      v8h lo = *(const v8h*)hp;
      v8h hi = *(const v8h*)(hp + 16);
#pragma unroll
      for (int i = 0; i < 8; ++i) { af[mt][i] = lo[i]; af[mt][i + 8] = hi[i]; }
    }
#pragma unroll
    for (int nt = 0; nt < 4; ++nt) {
      const _Float16* bp =
          W2h + (size_t)(colbase + nt * 16 + lm) * F_DIM + k0 + lk * 8;
      v8h lo = *(const v8h*)bp;
      v8h hi = *(const v8h*)(bp + 16);
#pragma unroll
      for (int i = 0; i < 8; ++i) { bf[nt][i] = lo[i]; bf[nt][i + 8] = hi[i]; }
    }

#pragma unroll
    for (int mt = 0; mt < 2; ++mt)
#pragma unroll
      for (int nt = 0; nt < 4; ++nt)
        acc[mt][nt] = __builtin_amdgcn_wmma_f32_16x16x32_f16(
            false, af[mt], false, bf[nt], (short)0, acc[mt][nt], false, false);
    // No trailing barrier: next iteration writes the OTHER buffer; re-use of
    // this buffer (i+2) is fenced by the barrier at i+1, and our LDS reads
    // complete (s_wait_dscnt) before the WMMAs that precede those writes.
  }

  // --- Store D + b2. D layout: VGPR v -> row v + 8*lk, col lm within tile.
#pragma unroll
  for (int nt = 0; nt < 4; ++nt) {
    int gn = colbase + nt * 16 + lm;
    float bias = b2[gn];
#pragma unroll
    for (int mt = 0; mt < 2; ++mt) {
      int gm = m0 + mt * 16 + 8 * lk;
#pragma unroll
      for (int v = 0; v < 8; ++v)
        out[(size_t)(gm + v) * E_DIM + gn] = acc[mt][nt][v] + bias;
    }
  }
}

extern "C" void kernel_launch(void* const* d_in, const int* in_sizes, int n_in,
                              void* d_out, int out_size, void* d_ws, size_t ws_size,
                              hipStream_t stream) {
  const float* x     = (const float*)d_in[0];
  const float* theta = (const float*)d_in[1];
  const float* W1    = (const float*)d_in[2];
  const float* b1    = (const float*)d_in[3];
  const float* W2    = (const float*)d_in[4];
  const float* b2    = (const float*)d_in[5];
  float* out = (float*)d_out;

  _Float16* W2h = (_Float16*)d_ws;  // needs E_DIM*F_DIM*2 = 2 MB of scratch

  // 1M elements, 4 per thread
  w2_cvt_kernel<<<(E_DIM * F_DIM) / (4 * 256), 256, 0, stream>>>(W2, W2h);
  // 512 workgroups x 256 threads; each owns 32 rows x all 512 cols
  ffq_fused_kernel<<<M_ROWS / 32, 256, 0, stream>>>(x, theta, W1, b1, b2, W2h, out);
}
